// DynamicLSTMRegressor_20332375179323
// MI455X (gfx1250) — compile-verified, matrix operand order checked by reading, not run on hardware
//
#include <hip/hip_runtime.h>
#include <cstdint>
#include <cstddef>

// ---------------------------------------------------------------------------
// DynamicLSTMRegressor for MI455X (gfx1250, wave32, WMMA)
//
//  * 8 workgroups, one per 16-row batch tile (WMMA M=16). Batch rows are
//    independent -> no grid-wide sync; recurrence coupling handled with
//    workgroup barriers inside one WGP.
//  * 1024 threads = 32 waves. Wave w owns h-columns [w*16, w*16+16) of both
//    layers -> gate column tiles {g*32+w}. Cell state c in registers.
//  * Weights pre-packed once to bf16 WMMA B-fragment layout, L2-resident,
//    re-streamed every step via address-space(1) pointers laundered per
//    iteration (prevents LICM hoist + scratch spill, keeps global_load_b128).
//  * h1/h2 in LDS in WMMA A-fragment layout (16 chunks x 1KB).
//  * x tile for t+1 prefetched via Tensor Data Mover, s_wait_tensorcnt sync.
//  * Accumulators start at 0; gate biases folded into activation scalars
//    (avoids keeping/spilling splatted bias v8f across the 4096-step loop).
//  * y = h2 @ w_out^T + b computed inline each step by waves 0..3.
// ---------------------------------------------------------------------------

#define Bv   128
#define Tv   4096
#define DIN  64
#define Hv   512
#define G4   2048
#define DOUT 64
#define NTHREADS 1024

typedef __attribute__((ext_vector_type(16))) __bf16 v16bf;
typedef __attribute__((ext_vector_type(8)))  float  v8f;
typedef __attribute__((ext_vector_type(4)))  unsigned int u32x4;
typedef __attribute__((ext_vector_type(8)))  int i32x8;
typedef __attribute__((ext_vector_type(4)))  int i32x4;

// explicit global-address-space pointer types (keep global_load_* encoding
// even after pointer laundering breaks address-space inference)
typedef __attribute__((address_space(1))) const __bf16 gas_bf16;
typedef __attribute__((address_space(1))) const v16bf gas_v16bf;

#ifdef __AMDGCN__
#define HAS_TDM 1
#else
#define HAS_TDM 0
#endif

__device__ __forceinline__ v8f wmma_bf16(v16bf a, v16bf b, v8f c) {
#ifdef __AMDGCN__
  return __builtin_amdgcn_wmma_f32_16x16x32_bf16(false, a, false, b, (short)0, c,
                                                 false, false);
#else
  return c;
#endif
}

__device__ __forceinline__ v8f zero8() {
  v8f r;
#pragma unroll
  for (int i = 0; i < 8; ++i) r[i] = 0.0f;
  return r;
}

__device__ __forceinline__ float sigm(float v) {
  return 1.0f / (1.0f + __expf(-v));
}

// ---------------------------------------------------------------------------
// Weight packing: W[N,K] (row-major f32) -> bf16 B-fragment blocks.
// Block (tile n0/16, chunk k0/32) = 32 lanes x 16 bf16, lane-major, so a lane
// reads its whole fragment as 32 contiguous bytes.
// 16-bit fragment mapping of (row-in-tile n, k-in-chunk kk):
//   lane = n + ((kk & 15) >= 8 ? 16 : 0);  slot = (kk & 7) + (kk >= 16 ? 8 : 0)
// ---------------------------------------------------------------------------
__global__ void pack_b_kernel(const float* __restrict__ W,
                              unsigned short* __restrict__ outp,
                              int N, int K) {
  int idx = blockIdx.x * 256 + threadIdx.x;
  if (idx >= N * K) return;
  int n = idx / K, k = idx % K;
  int tile = n >> 4, n_in = n & 15;
  int c = k >> 5, kk = k & 31;
  int lane = n_in + (((kk & 15) >= 8) ? 16 : 0);
  int slot = (kk & 7) + ((kk >= 16) ? 8 : 0);
  __bf16 h = (__bf16)W[idx];
  outp[((size_t)tile * (size_t)(K >> 5) + (size_t)c) * 512 + lane * 16 + slot] =
      __builtin_bit_cast(unsigned short, h);
}

// ---------------------------------------------------------------------------
// TDM: async 2D tile load (16 rows x 64 f32, row stride T*64 f32) -> LDS.
// D# per CDNA5 ISA.
// ---------------------------------------------------------------------------
__device__ __forceinline__ void tdm_load_x_tile(const float* gsrc,
                                                unsigned lds_off) {
#if HAS_TDM
  unsigned long long ga = (unsigned long long)(uintptr_t)gsrc;
  u32x4 g0;
  g0[0] = 1u;                                   // count=1, user descriptor
  g0[1] = lds_off;                              // LDS byte offset
  g0[2] = (unsigned)(ga & 0xffffffffu);         // global_addr[31:0]
  g0[3] = (unsigned)((ga >> 32) & 0x1ffffffu)   // global_addr[56:32]
          | (2u << 30);                         // type=2 ("image")
  i32x8 g1;
  g1[0] = (int)(2u << 16);                      // data_size = 4B (code 2)
  g1[1] = (int)(64u << 16);                     // tensor_dim0 = 64 (lo16)
  g1[2] = 0;                                    // dim0 hi / dim1 lo16 = 0
  g1[3] = (int)(0x8u | (64u << 16));            // dim1 = 0x80000 rows; tile_dim0=64
  g1[4] = 16;                                   // tile_dim1 = 16 rows
  g1[5] = (int)(Tv * 64);                       // tensor_dim0_stride = T*64 elems
  g1[6] = 0;
  g1[7] = 0;
  i32x4 z4 = {0, 0, 0, 0};
#if __clang_major__ >= 23
  i32x8 z8 = {0, 0, 0, 0, 0, 0, 0, 0};
  __builtin_amdgcn_tensor_load_to_lds(g0, g1, z4, z4, z8, 0);
#else
  __builtin_amdgcn_tensor_load_to_lds(g0, g1, z4, z4, 0);
#endif
#else
  (void)gsrc; (void)lds_off;
#endif
}

// ---------------------------------------------------------------------------
// One GEMM slice: acc[g] += A(16xK from frag buffer) x B(tile g*32+w)
// ---------------------------------------------------------------------------
template <int NC>
__device__ __forceinline__ void gemm4(v8f acc[4], const __bf16* af,
                                      gas_bf16* wblk, int w, int lane) {
  const __bf16* al = af + lane * 16;
  gas_bf16* wl = wblk + lane * 16;
  for (int c = 0; c < NC; ++c) {
    v16bf a = *(const v16bf*)(al + c * 512);
#pragma unroll
    for (int g = 0; g < 4; ++g) {
      int tile = g * 32 + w;
      v16bf b = *(gas_v16bf*)(wl + ((size_t)tile * NC + (size_t)c) * 512);
      acc[g] = wmma_bf16(a, b, acc[g]);
    }
  }
}

// gates (i,f,g,o) + biases -> new cell state + h values (accumulator layout:
// lane l, reg r holds element (m = r + 8*(l>=16), n = l&15) of each tile)
__device__ __forceinline__ void lstm_act(const v8f acc[4], const float bv[4],
                                         v8f& cs, float hv[8]) {
#pragma unroll
  for (int r = 0; r < 8; ++r) {
    float iv = sigm(acc[0][r] + bv[0]);
    float fv = sigm(acc[1][r] + bv[1]);
    float gv = tanhf(acc[2][r] + bv[2]);
    float ov = sigm(acc[3][r] + bv[3]);
    float cn = fv * cs[r] + iv * gv;
    cs[r] = cn;
    hv[r] = ov * tanhf(cn);
  }
}

// Scatter this wave's 16 h-columns into the A-fragment-layout h buffer.
// Wave w owns k-chunk w>>1, half s=w&1 of it.
__device__ __forceinline__ void store_h(__bf16* hfrag, const float hv[8],
                                        int w, int lane) {
  int nloc = lane & 15;
  int mb = (lane >> 4) << 3;
  int c = w >> 1, s = w & 1;
  int jp = (nloc & 7) + (s << 3);
  int lpb = (nloc >= 8) ? 16 : 0;
  __bf16* base = hfrag + c * 512;
#pragma unroll
  for (int r = 0; r < 8; ++r) {
    int lp = mb + r + lpb;
    base[lp * 16 + jp] = (__bf16)hv[r];
  }
}

// ---------------------------------------------------------------------------
// Main persistent kernel: one workgroup per 16-row batch tile.
// ---------------------------------------------------------------------------
__global__ __launch_bounds__(NTHREADS, 1) void lstm_kernel(
    const float* __restrict__ x,
    const float* __restrict__ b_ih1, const float* __restrict__ b_hh1,
    const float* __restrict__ b_ih2, const float* __restrict__ b_hh2,
    const float* __restrict__ b_out,
    const unsigned short* __restrict__ pih1,
    const unsigned short* __restrict__ phh1,
    const unsigned short* __restrict__ pih2,
    const unsigned short* __restrict__ phh2,
    const unsigned short* __restrict__ pout,
    float* __restrict__ out) {
  __shared__ __align__(32) __bf16 h1frag[16 * 512];  // 16 k-chunks, A-frag layout
  __shared__ __align__(32) __bf16 h2frag[16 * 512];
  __shared__ __align__(32) __bf16 xfrag[2 * 512];    // 2 k-chunks of x tile
  __shared__ __align__(16) float xraw[2][16 * 64];   // TDM double buffer

  const int tid = threadIdx.x;
  const int w = tid >> 5;         // wave 0..31
  const int lane = tid & 31;
  const int nloc = lane & 15;
  const int mbase = (lane >> 4) << 3;
  const int b0 = blockIdx.x << 4; // batch tile start

  // packed-weight bases kept as integers so per-iteration laundering cannot
  // demote the loads to flat addressing
  uintptr_t a_ih1 = (uintptr_t)pih1;
  uintptr_t a_hh1 = (uintptr_t)phh1;
  uintptr_t a_ih2 = (uintptr_t)pih2;
  uintptr_t a_hh2 = (uintptr_t)phh2;
  uintptr_t a_out = (uintptr_t)pout;

  // h0 = c0 = 0
  for (int i = tid; i < 16 * 512; i += NTHREADS) {
    h1frag[i] = (__bf16)0.0f;
    h2frag[i] = (__bf16)0.0f;
  }
  v8f c1 = zero8();
  v8f c2 = zero8();

  // combined gate biases for this wave's columns (scalar per gate)
  float b1v[4], b2v[4];
#pragma unroll
  for (int g = 0; g < 4; ++g) {
    int col = g * Hv + w * 16 + nloc;
    b1v[g] = b_ih1[col] + b_hh1[col];
    b2v[g] = b_ih2[col] + b_hh2[col];
  }
  float byv = (w < 4) ? b_out[w * 16 + nloc] : 0.0f;

#if HAS_TDM
  if (w == 0) {  // prefetch x tile for t=0
    tdm_load_x_tile(x + (size_t)b0 * Tv * 64,
                    (unsigned)(uintptr_t)&xraw[0][0]);
  }
#endif

  for (int t = 0; t < Tv; ++t) {
    // Launder the packed-weight bases once per timestep so LICM/GVN cannot
    // hoist the (t-invariant) B-fragment loads out of the 4096-step loop and
    // spill them to scratch; weights re-stream from L2 each step.
#ifdef __AMDGCN__
    asm volatile("" : "+s"(a_ih1), "+s"(a_hh1), "+s"(a_ih2), "+s"(a_hh2),
                      "+s"(a_out));
#endif
    gas_bf16* bih1 = (gas_bf16*)a_ih1;
    gas_bf16* bhh1 = (gas_bf16*)a_hh1;
    gas_bf16* bih2 = (gas_bf16*)a_ih2;
    gas_bf16* bhh2 = (gas_bf16*)a_hh2;
    gas_bf16* bout = (gas_bf16*)a_out;

#if HAS_TDM
    if (w == 0) __builtin_amdgcn_s_wait_tensorcnt(0);
#endif
    __syncthreads();
#if HAS_TDM
    if (w == 0 && t + 1 < Tv) {  // prefetch next step's x tile
      tdm_load_x_tile(x + ((size_t)b0 * Tv + (size_t)(t + 1)) * 64,
                      (unsigned)(uintptr_t)&xraw[(t + 1) & 1][0]);
    }
#endif
    // convert current x tile (f32) -> bf16 A-fragment layout
    for (int e = tid; e < 16 * 64; e += NTHREADS) {
      int m = e >> 6, k = e & 63;
#if HAS_TDM
      float xv = xraw[t & 1][e];
#else
      float xv = x[((size_t)(b0 + m) * Tv + (size_t)t) * 64 + k];
#endif
      int c = k >> 5, kk = k & 31;
      int lp = m + (((kk & 15) >= 8) ? 16 : 0);
      int jp = (kk & 7) + ((kk >= 16) ? 8 : 0);
      xfrag[c * 512 + lp * 16 + jp] = (__bf16)xv;
    }
    __syncthreads();

    // ---- Layer 1: gates = x@Wih1^T + h1@Whh1^T (+ bias in activation) ----
    v8f acc[4];
#pragma unroll
    for (int g = 0; g < 4; ++g) acc[g] = zero8();
    gemm4<2>(acc, xfrag, bih1, w, lane);
    gemm4<16>(acc, h1frag, bhh1, w, lane);
    float hv1[8];
    lstm_act(acc, b1v, c1, hv1);
    __syncthreads();                 // all reads of h1frag done
    store_h(h1frag, hv1, w, lane);
    __syncthreads();

    // ---- Layer 2: gates = h1@Wih2^T + h2@Whh2^T (+ bias in activation) ----
#pragma unroll
    for (int g = 0; g < 4; ++g) acc[g] = zero8();
    gemm4<16>(acc, h1frag, bih2, w, lane);
    gemm4<16>(acc, h2frag, bhh2, w, lane);
    float hv2[8];
    lstm_act(acc, b2v, c2, hv2);
    __syncthreads();                 // all reads of h2frag done
    store_h(h2frag, hv2, w, lane);
    __syncthreads();

    // ---- Output projection: y[t] = h2 @ Wout^T + b_out (waves 0..3) ----
    if (w < 4) {
      v8f ya = zero8();
      const __bf16* hl = h2frag + lane * 16;
      gas_bf16* wl = bout + (size_t)w * 16 * 512 + lane * 16;
      for (int c = 0; c < 16; ++c) {
        v16bf a = *(const v16bf*)(hl + c * 512);
        v16bf b = *(gas_v16bf*)(wl + (size_t)c * 512);
        ya = wmma_bf16(a, b, ya);
      }
      int colg = (w << 4) + nloc;
#pragma unroll
      for (int r = 0; r < 8; ++r) {
        out[((size_t)(b0 + mbase + r) * Tv + (size_t)t) * 64 + colg] =
            ya[r] + byv;
      }
    }
    // next iteration's pre-write barriers protect h2frag vs. these reads
  }
}

// ---------------------------------------------------------------------------
extern "C" void kernel_launch(void* const* d_in, const int* in_sizes, int n_in,
                              void* d_out, int out_size, void* d_ws,
                              size_t ws_size, hipStream_t stream) {
  const float* x     = (const float*)d_in[0];
  const float* w_ih1 = (const float*)d_in[1];
  const float* w_hh1 = (const float*)d_in[2];
  const float* b_ih1 = (const float*)d_in[3];
  const float* b_hh1 = (const float*)d_in[4];
  const float* w_ih2 = (const float*)d_in[5];
  const float* w_hh2 = (const float*)d_in[6];
  const float* b_ih2 = (const float*)d_in[7];
  const float* b_hh2 = (const float*)d_in[8];
  const float* w_out = (const float*)d_in[9];
  const float* b_out = (const float*)d_in[10];
  float* out = (float*)d_out;

  // workspace: packed bf16 weights (~6.4 MB)
  unsigned short* ws = (unsigned short*)d_ws;
  unsigned short* blk_ih1 = ws;
  unsigned short* blk_hh1 = blk_ih1 + (size_t)G4 * DIN;
  unsigned short* blk_ih2 = blk_hh1 + (size_t)G4 * Hv;
  unsigned short* blk_hh2 = blk_ih2 + (size_t)G4 * Hv;
  unsigned short* blk_out = blk_hh2 + (size_t)G4 * Hv;

  pack_b_kernel<<<dim3((G4 * DIN + 255) / 256), dim3(256), 0, stream>>>(
      w_ih1, blk_ih1, G4, DIN);
  pack_b_kernel<<<dim3((G4 * Hv + 255) / 256), dim3(256), 0, stream>>>(
      w_hh1, blk_hh1, G4, Hv);
  pack_b_kernel<<<dim3((G4 * Hv + 255) / 256), dim3(256), 0, stream>>>(
      w_ih2, blk_ih2, G4, Hv);
  pack_b_kernel<<<dim3((G4 * Hv + 255) / 256), dim3(256), 0, stream>>>(
      w_hh2, blk_hh2, G4, Hv);
  pack_b_kernel<<<dim3((DOUT * Hv + 255) / 256), dim3(256), 0, stream>>>(
      w_out, blk_out, DOUT, Hv);

  lstm_kernel<<<dim3(Bv / 16), dim3(NTHREADS), 0, stream>>>(
      x, b_ih1, b_hh1, b_ih2, b_hh2, b_out, blk_ih1, blk_hh1, blk_ih2, blk_hh2,
      blk_out, out);
}
